// TtFalconAttention_27049704030621
// MI455X (gfx1250) — compile-verified, matrix operand order checked by reading, not run on hardware
//
#include <hip/hip_runtime.h>
#include <hip/hip_bf16.h>

// ---------------------------------------------------------------------------
// Falcon MQA attention layer for MI455X (gfx1250), bf16 WMMA + f32 accumulate.
// HBM-bound (~210 MB @ 23.3 TB/s ~ 9us; math ~105 GFLOP << matrix roof):
// weights stream fp32 once (converted to bf16 in registers during staging),
// activations are pre-converted to bf16 once so GEMM A-tiles and attention
// K/V tiles move via GLOBAL_LOAD_ASYNC_TO_LDS_B128 (ASYNCcnt), feeding
// v_wmma_f32_16x16x32_bf16. GEMM dims are compile-time so strided weight
// loads fold into immediate VMEM offsets.
// ---------------------------------------------------------------------------

typedef __bf16 bf16_t;
typedef __attribute__((ext_vector_type(16))) __bf16 v16bf;
typedef __attribute__((ext_vector_type(8)))  __bf16 v8bf;
typedef __attribute__((ext_vector_type(4)))  __bf16 v4bf;
typedef __attribute__((ext_vector_type(8)))  float  v8f;
typedef __attribute__((ext_vector_type(4)))  int    vi4;

#define HIDDEN 4544
#define NH     71
#define HD     64
#define SEQ    1024
#define NQKV   4672   // (NH+2)*HD

#if defined(__has_builtin)
#if __has_builtin(__builtin_amdgcn_global_load_async_to_lds_b128)
#define HAVE_ASYNC_LDS 1
#endif
#endif

#define AS_GLOBAL __attribute__((address_space(1)))
#define AS_LOCAL  __attribute__((address_space(3)))

static __device__ __forceinline__ void async_copy16(const void* g, void* l) {
#if defined(HAVE_ASYNC_LDS)
  __builtin_amdgcn_global_load_async_to_lds_b128(
      (AS_GLOBAL vi4*)g, (AS_LOCAL vi4*)l, 0, 0);
#else
  *(uint4*)l = *(const uint4*)g;
#endif
}

static __device__ __forceinline__ void async_wait_all() {
#if defined(HAVE_ASYNC_LDS)
#if __has_builtin(__builtin_amdgcn_s_wait_asynccnt)
  __builtin_amdgcn_s_wait_asynccnt(0);
#else
  asm volatile("s_wait_asynccnt 0" ::: "memory");
#endif
#endif
}

// Per-lane WMMA 16x16x32 A/B fragment: two contiguous 16B chunks at kbase and
// kbase+16 (ISA 16-bit A layout; B mirrored when stored [n][k] row-major).
static __device__ __forceinline__ v16bf make_frag(const bf16_t* p) {
  v8bf lo = *(const v8bf*)(p);
  v8bf hi = *(const v8bf*)(p + 16);
  return __builtin_shufflevector(lo, hi,
      0, 1, 2, 3, 4, 5, 6, 7, 8, 9, 10, 11, 12, 13, 14, 15);
}

// ---------------------------------------------------------------------------
// fp32 -> bf16 pack (activations): one float4 -> v4bf per thread.
// ---------------------------------------------------------------------------
__global__ __launch_bounds__(256)
void cvt_f32_bf16(const float* __restrict__ src, bf16_t* __restrict__ dst,
                  int n4) {
  int i = blockIdx.x * 256 + threadIdx.x;
  if (i >= n4) return;
  float4 v = ((const float4*)src)[i];
  v4bf p;
  p[0] = (bf16_t)v.x; p[1] = (bf16_t)v.y;
  p[2] = (bf16_t)v.z; p[3] = (bf16_t)v.w;
  ((v4bf*)dst)[i] = p;
}

// ---------------------------------------------------------------------------
// GEMM: C[M,N] = A[M,K](bf16) * B[K,N](fp32), f32 accumulate via bf16 WMMA.
// Block = 256 threads (8 waves), tile 128x64, BK=32; wave = 2x2 16x16 tiles.
// A tile staged with async global->LDS b128 copies; B tile converted in
// registers (loads first with compile-time strides -> immediate offsets,
// then cvt+store, so loads pipeline with no address-add chains).
// ---------------------------------------------------------------------------
template <int N, int K>
__global__ __launch_bounds__(256)
void gemm_abf16_wmma(const bf16_t* __restrict__ A, const float* __restrict__ B,
                     float* __restrict__ C) {
  __shared__ bf16_t As[128][32];  // [m][k]
  __shared__ bf16_t Bs[64][32];   // [n][k]

  const int tid  = threadIdx.x;
  const int lane = tid & 31;
  const int wave = tid >> 5;
  const int wr   = (wave >> 1) * 32;
  const int wc   = (wave & 1) * 32;
  const int m0   = blockIdx.y * 128;
  const int n0   = blockIdx.x * 64;
  const int l16  = lane & 15;
  const int kb   = (lane >> 4) * 8;
  const int hi8  = (lane >> 4) * 8;

  // staging task coordinates
  const int ar = tid >> 2, ac = (tid & 3) * 8;    // A chunk: row, col(8 bf16)
  const int bn = tid & 63, bk4 = (tid >> 6) * 4;  // B task: n, k-quad

  v8f acc[2][2];
#pragma unroll
  for (int i = 0; i < 2; ++i)
#pragma unroll
    for (int j = 0; j < 2; ++j) acc[i][j] = v8f{};

  for (int k0 = 0; k0 < K; k0 += 32) {
    // --- A tile: 128x32 bf16 = 512 x 16B chunks, async copy, 2/thread.
#pragma unroll
    for (int t = 0; t < 2; ++t) {
      int r = ar + t * 64;
      async_copy16(&A[(size_t)(m0 + r) * K + k0 + ac], &As[r][ac]);
    }
    // --- B tile: 32x64 fp32 -> Bs[n][k]. All loads issued first; k-stride
    // offsets are compile-time constants -> immediate VMEM offsets.
    const float* bp = B + (size_t)(k0 + bk4) * N + n0 + bn;
    float b0[4], b1[4];
#pragma unroll
    for (int i = 0; i < 4; ++i) b0[i] = bp[i * N];
#pragma unroll
    for (int i = 0; i < 4; ++i) b1[i] = bp[(16 + i) * N];
    v4bf p0, p1;
#pragma unroll
    for (int i = 0; i < 4; ++i) { p0[i] = (bf16_t)b0[i]; p1[i] = (bf16_t)b1[i]; }
    *(v4bf*)&Bs[bn][bk4]      = p0;
    *(v4bf*)&Bs[bn][bk4 + 16] = p1;

    async_wait_all();
    __syncthreads();

    // Prefetch next K-tile (global_prefetch_b8)
    if (k0 + 32 < K) {
      {
        int r = tid >> 1, c = (tid & 1) * 16;
        __builtin_prefetch(&A[(size_t)(m0 + r) * K + (k0 + 32 + c)], 0, 3);
      }
      {
        int kk = tid >> 3, n = (tid & 7) * 8;
        __builtin_prefetch(&B[(size_t)(k0 + 32 + kk) * N + (n0 + n)], 0, 3);
      }
    }

    v16bf afrag[2], bfrag[2];
#pragma unroll
    for (int t = 0; t < 2; ++t) {
      afrag[t] = make_frag(&As[wr + t * 16 + l16][kb]);
      bfrag[t] = make_frag(&Bs[wc + t * 16 + l16][kb]);
    }
#pragma unroll
    for (int tr = 0; tr < 2; ++tr)
#pragma unroll
      for (int tc = 0; tc < 2; ++tc)
        acc[tr][tc] = __builtin_amdgcn_wmma_f32_16x16x32_bf16(
            false, afrag[tr], false, bfrag[tc], (short)0, acc[tr][tc],
            false, false);
    __syncthreads();
  }

  // Epilogue: D layout: VGPR vr -> M=vr (lanes 0-15) / M=vr+8 (lanes 16-31)
#pragma unroll
  for (int tr = 0; tr < 2; ++tr)
#pragma unroll
    for (int tc = 0; tc < 2; ++tc) {
      int col = n0 + wc + tc * 16 + l16;
#pragma unroll
      for (int vr = 0; vr < 8; ++vr) {
        int row = m0 + wr + tr * 16 + vr + hi8;
        C[(size_t)row * N + col] = acc[tr][tc][vr];
      }
    }
}

// ---------------------------------------------------------------------------
// RoPE on q/k + bf16 pack. Q head-major [NH][S][HD]; K [S][HD]; V TRANSPOSED
// [HD][S] so attention V-tiles are contiguous copies.
// head-slot: 0..70 = q heads, 71 = shared k, 72 = shared v.
// ---------------------------------------------------------------------------
__global__ __launch_bounds__(256)
void rope_pack(const float* __restrict__ fused, bf16_t* __restrict__ Qb,
               bf16_t* __restrict__ Kb, bf16_t* __restrict__ VbT) {
  int gid = blockIdx.x * 256 + threadIdx.x;
  int j = gid & 31;
  int h = (gid >> 5) % 73;
  int s = gid / (32 * 73);
  if (s >= SEQ) return;

  const float* row = fused + (size_t)s * NQKV;
  if (h < 72) {
    int off = (h < NH) ? h * HD : NH * HD;
    float x1 = row[off + j];
    float x2 = row[off + j + 32];
    float invf = __powf(10000.0f, -((float)j) / 32.0f);
    float fr = (float)s * invf;
    float sn, cs;
    __sincosf(fr, &sn, &cs);
    float o1 = x1 * cs - x2 * sn;
    float o2 = x2 * cs + x1 * sn;
    if (h < NH) {
      bf16_t* q = Qb + ((size_t)h * SEQ + s) * HD;
      q[j]      = (bf16_t)o1;
      q[j + 32] = (bf16_t)o2;
    } else {
      Kb[(size_t)s * HD + j]      = (bf16_t)o1;
      Kb[(size_t)s * HD + j + 32] = (bf16_t)o2;
    }
  } else {
    VbT[(size_t)j * SEQ + s]        = (bf16_t)row[(NH + 1) * HD + j];
    VbT[(size_t)(j + 32) * SEQ + s] = (bf16_t)row[(NH + 1) * HD + j + 32];
  }
}

// ---------------------------------------------------------------------------
// Flash attention: one block per (64 q-rows, head). 4 waves x 16 rows.
// K tile [key][d] and V^T tile [d][key] staged via async global->LDS b128;
// online softmax with xor-shuffles over the 16-lane N-group of the WMMA D
// layout. Output written as bf16 (consumed by dense GEMM).
// ---------------------------------------------------------------------------
__global__ __launch_bounds__(128)
void flash_attn(const bf16_t* __restrict__ Qb, const bf16_t* __restrict__ Kb,
                const bf16_t* __restrict__ VbT, bf16_t* __restrict__ Mrg) {
  __shared__ bf16_t Ks[64][64];     // [key][d]
  __shared__ bf16_t VT[64][64];     // [d][key]
  __shared__ bf16_t Pb[4][16][64];  // per-wave probs staging [row][key]

  const int tid  = threadIdx.x;
  const int lane = tid & 31;
  const int wave = tid >> 5;
  const int h    = blockIdx.y;
  const int q0   = blockIdx.x * 64;
  const int l16  = lane & 15;
  const int hi8  = (lane >> 4) * 8;
  const int kb   = (lane >> 4) * 8;

  // Q fragments for this wave's 16 rows (d-chunks 0..31, 32..63)
  const int qrow = q0 + wave * 16 + l16;
  const bf16_t* qp = Qb + ((size_t)h * SEQ + qrow) * HD;
  v16bf aq[2];
  aq[0] = make_frag(qp + 0 + kb);
  aq[1] = make_frag(qp + 32 + kb);

  v8f o[4];
#pragma unroll
  for (int t = 0; t < 4; ++t) o[t] = v8f{};
  float rm[8], rl[8];
#pragma unroll
  for (int vr = 0; vr < 8; ++vr) { rm[vr] = -1e30f; rl[vr] = 0.0f; }

  const int nkb = blockIdx.x + 1;  // causal
  for (int kblk = 0; kblk < nkb; ++kblk) {
    const int ks0 = kblk * 64;
    // ---- K tile: contiguous 8KB; V^T tile: 64 rows x 128B. Async copies.
    {
      const char* gk = (const char*)(Kb + (size_t)ks0 * HD);
      char* lk = (char*)&Ks[0][0];
#pragma unroll
      for (int t = 0; t < 4; ++t) {               // 512 x 16B chunks
        int e = (tid + t * 128) * 16;
        async_copy16(gk + e, lk + e);
      }
#pragma unroll
      for (int t = 0; t < 4; ++t) {               // 512 chunks: d row, 8/row
        int e = tid + t * 128;
        int d = e >> 3, c = (e & 7) * 8;          // 8 bf16 = 16B per chunk
        async_copy16(VbT + (size_t)d * SEQ + ks0 + c, &VT[d][c]);
      }
      async_wait_all();
    }
    __syncthreads();

    // ---- S = Q K^T * scale + causal
    float st[4][8];
#pragma unroll
    for (int tc = 0; tc < 4; ++tc) {
      v8f s8 = v8f{};
#pragma unroll
      for (int kc = 0; kc < 2; ++kc) {
        v16bf bf = make_frag(&Ks[tc * 16 + l16][kc * 32 + kb]);
        s8 = __builtin_amdgcn_wmma_f32_16x16x32_bf16(
            false, aq[kc], false, bf, (short)0, s8, false, false);
      }
      int col = ks0 + tc * 16 + l16;
#pragma unroll
      for (int vr = 0; vr < 8; ++vr) {
        int row = q0 + wave * 16 + vr + hi8;
        st[tc][vr] = s8[vr] * 0.125f + ((col <= row) ? 0.0f : -1.0e9f);
      }
    }

    // ---- Online softmax (row stats per (vr, lane-half))
    float nm[8], alpha[8];
#pragma unroll
    for (int vr = 0; vr < 8; ++vr) {
      float mx = fmaxf(fmaxf(st[0][vr], st[1][vr]),
                       fmaxf(st[2][vr], st[3][vr]));
#pragma unroll
      for (int off = 1; off < 16; off <<= 1)
        mx = fmaxf(mx, __shfl_xor(mx, off, 32));
      nm[vr] = fmaxf(rm[vr], mx);
      alpha[vr] = __expf(rm[vr] - nm[vr]);
      rm[vr] = nm[vr];
    }
#pragma unroll
    for (int tc = 0; tc < 4; ++tc)
#pragma unroll
      for (int vr = 0; vr < 8; ++vr)
        st[tc][vr] = __expf(st[tc][vr] - nm[vr]);
#pragma unroll
    for (int vr = 0; vr < 8; ++vr) {
      float sm = st[0][vr] + st[1][vr] + st[2][vr] + st[3][vr];
#pragma unroll
      for (int off = 1; off < 16; off <<= 1)
        sm += __shfl_xor(sm, off, 32);
      rl[vr] = rl[vr] * alpha[vr] + sm;
    }

    // ---- Rescale O, stage P to LDS (D layout -> row-major for A frags)
#pragma unroll
    for (int tc = 0; tc < 4; ++tc)
#pragma unroll
      for (int vr = 0; vr < 8; ++vr) {
        o[tc][vr] *= alpha[vr];
        Pb[wave][vr + hi8][tc * 16 + l16] = (bf16_t)st[tc][vr];
      }

    // ---- O += P * V
#pragma unroll
    for (int kc = 0; kc < 2; ++kc) {
      v16bf ap = make_frag(&Pb[wave][l16][kc * 32 + kb]);
#pragma unroll
      for (int tc = 0; tc < 4; ++tc) {
        v16bf bv = make_frag(&VT[tc * 16 + l16][kc * 32 + kb]);
        o[tc] = __builtin_amdgcn_wmma_f32_16x16x32_bf16(
            false, ap, false, bv, (short)0, o[tc], false, false);
      }
    }
    __syncthreads();
  }

  // ---- merged[s][h*64+d] = bf16(O / l)
#pragma unroll
  for (int vr = 0; vr < 8; ++vr) {
    float inv = 1.0f / rl[vr];
    int row = q0 + wave * 16 + vr + hi8;
#pragma unroll
    for (int tc = 0; tc < 4; ++tc)
      Mrg[(size_t)row * HIDDEN + h * HD + tc * 16 + l16] =
          (bf16_t)(o[tc][vr] * inv);
  }
}

// ---------------------------------------------------------------------------
// cvt -> qkv GEMM -> rope/pack -> flash attention -> dense GEMM
// ---------------------------------------------------------------------------
extern "C" void kernel_launch(void* const* d_in, const int* in_sizes, int n_in,
                              void* d_out, int out_size, void* d_ws, size_t ws_size,
                              hipStream_t stream) {
  const float* x      = (const float*)d_in[0];  // [1,1,1024,4544]
  // d_in[1] = attention_mask (causal; reproduced analytically)
  const float* wqkv   = (const float*)d_in[2];  // [4544, 4672]
  const float* wdense = (const float*)d_in[3];  // [4544, 4544]
  float* out = (float*)d_out;                   // [1,1,1024,4544]

  char* ws = (char*)d_ws;
  size_t off = 0;
  bf16_t* xb     = (bf16_t*)(ws + off); off += (size_t)SEQ * HIDDEN * sizeof(bf16_t);
  float*  fused  = (float*)(ws + off);  off += (size_t)SEQ * NQKV * sizeof(float);
  bf16_t* Qb     = (bf16_t*)(ws + off); off += (size_t)NH * SEQ * HD * sizeof(bf16_t);
  bf16_t* Kb     = (bf16_t*)(ws + off); off += (size_t)SEQ * HD * sizeof(bf16_t);
  bf16_t* VbT    = (bf16_t*)(ws + off); off += (size_t)HD * SEQ * sizeof(bf16_t);
  bf16_t* merged = (bf16_t*)(ws + off); off += (size_t)SEQ * HIDDEN * sizeof(bf16_t);

  // 0) x -> bf16 (one cheap pass; halves L2 traffic for re-read activations)
  cvt_f32_bf16<<<(SEQ * HIDDEN / 4 + 255) / 256, 256, 0, stream>>>(
      x, xb, SEQ * HIDDEN / 4);

  // 1) fused = xb @ Wqkv   (1024 x 4544 x 4672)
  dim3 g1(NQKV / 64, SEQ / 128);
  gemm_abf16_wmma<NQKV, HIDDEN><<<g1, 256, 0, stream>>>(xb, wqkv, fused);

  // 2) RoPE + pack to bf16 (V transposed)
  rope_pack<<<(SEQ * 73 * 32) / 256, 256, 0, stream>>>(fused, Qb, Kb, VbT);

  // 3) flash attention -> merged bf16
  dim3 g3(SEQ / 64, NH);
  flash_attn<<<g3, 128, 0, stream>>>(Qb, Kb, VbT, merged);

  // 4) out = merged @ Wdense  (1024 x 4544 x 4544)
  dim3 g4(HIDDEN / 64, SEQ / 128);
  gemm_abf16_wmma<HIDDEN, HIDDEN><<<g4, 256, 0, stream>>>(merged, wdense, out);
}
